// HSSLinearAttention_62302795596509
// MI455X (gfx1250) — matrix operand order repaired and checked by reading, not compiled, not run on hardware
//
#include <hip/hip_runtime.h>

// ---------------------------------------------------------------------------
// HSS linear attention forward for MI455X (gfx1250, wave32, WMMA).
//
// Pipeline (all on `stream`, stream-ordered):
//   1. packKernel   : Wq|Wk|Wv fp32 -> bf16 WMMA B-fragment layout in ws
//   2. qkvKernel    : Q=elu1(xWq), K=elu1(xWk), V=xWv via v_wmma_f32_16x16x32_bf16
//   3. deltaKernel  : per (batch,chunk) additive state delta (896 floats)
//   4. prefixKernel : exclusive prefix-sum of chunk states (scan parallelization)
//   5. scanKernel   : exact per-chunk replay of the recurrence from prefix state
//   6. outKernel    : Y @ Wo via v_wmma_f32_16x16x4_f32 (exact fp32)
// ---------------------------------------------------------------------------

typedef __attribute__((ext_vector_type(16))) __bf16        v16bf;
typedef __attribute__((ext_vector_type(8)))  float         v8f;
typedef __attribute__((ext_vector_type(2)))  float         v2f;
typedef __attribute__((ext_vector_type(8)))  unsigned int  v8u;

#define BATCH     8
#define TLEN      2048
#define DMODEL    1024
#define DH        64
#define CHUNK     64
#define NCHUNK    32          // TLEN / CHUNK
#define STATE_N   896         // 256 leaf + 576 Ws + 64 z
#define NSLOT     88          // per-level projection slots: 32+32+16+8

// ---- structure helpers (pure arithmetic, no memory) -----------------------
__device__ __forceinline__ int BSf(int l)        { return 4 << l; }               // 4,8,16,32
__device__ __forceinline__ int RLf(int l)        { return l ? 8 : 4; }            // 4,8,8,8
__device__ __forceinline__ int slotOff(int l)    { return l==0?0 : l==1?32 : l==2?64 : 80; }
__device__ __forceinline__ int baseOff(int l)    { return l==0?0 : l==1?16 : l==2?80 : 208; } // bs*rl cumsum (total 464)
__device__ __forceinline__ int wOff(int l)       { return l==0?256 : l==1?384 : l==2?640 : 768; }
__device__ __forceinline__ int levelOfSlot(int s){ return s<32?0 : s<64?1 : s<80?2 : 3; }
__device__ __forceinline__ int levelOfBase(int p){ return p<16?0 : p<80?1 : p<208?2 : 3; }
__device__ __forceinline__ int wLevel(int e)     { return e<384?0 : e<640?1 : e<768?2 : 3; }

// branchless elu(x)+1: x>0 ? x+1 : exp(x)   (exp on clamped arg, then select)
__device__ __forceinline__ float elu1(float v) {
  float ex = __expf(fminf(v, 0.f));
  return (v > 0.f) ? (v + 1.f) : ex;
}

struct Bases { const float* p[16]; };  // UL0-3, VR0-3, UR0-3, VL0-3

// ---------------------------------------------------------------------------
// 1. Pack Wq|Wk|Wv into bf16 B-fragment lane order.
//    Fragment (tile,kf): 32x16 bf16; lane l holds col = l&15,
//    K = kf*32 + (l<16?0:8) + j + (j<8?0:8), j=0..15 packed as 16 bf16 (8 dwords).
// ---------------------------------------------------------------------------
__global__ void hss_packKernel(const float* __restrict__ Wq,
                               const float* __restrict__ Wk,
                               const float* __restrict__ Wv,
                               v8u* __restrict__ wpack) {
  int t = blockIdx.x * 256 + threadIdx.x;
  if (t >= 12 * 32 * 32) return;
  int lane = t & 31, kf = (t >> 5) & 31, tile = t >> 10;
  int mat = tile >> 2, ntile = tile & 3;
  const float* W = (mat == 0) ? Wq : (mat == 1) ? Wk : Wv;
  int col = ntile * 16 + (lane & 15);
  int kb  = kf * 32 + ((lane < 16) ? 0 : 8);
  v16bf frag;
#pragma unroll
  for (int j = 0; j < 16; ++j)
    frag[j] = (__bf16)W[(size_t)(kb + j + (j < 8 ? 0 : 8)) * DH + col];
  wpack[t] = __builtin_bit_cast(v8u, frag);
}

// ---------------------------------------------------------------------------
// 2. QKV projection: one wave computes one 16x16 output tile, K-loop of 32
//    v_wmma_f32_16x16x32_bf16. grid=(1024, 12): y = {q0..3,k0..3,v0..3}.
// ---------------------------------------------------------------------------
__global__ void hss_qkvKernel(const float* __restrict__ x,
                              const v8u* __restrict__ wpack,
                              float* __restrict__ Qb, float* __restrict__ Kb,
                              float* __restrict__ Vb) {
  int lane  = threadIdx.x;           // 32-thread block: one wave
  int mtile = blockIdx.x;            // 0..1023
  int tile  = blockIdx.y;            // 0..11
  int mat   = tile >> 2, ntile = tile & 3;
  int row   = mtile * 16 + (lane & 15);
  const float* xr = x + (size_t)row * DMODEL;
  int c0sel = (lane < 16) ? 0 : 8;
  v8f acc = {};
#pragma unroll 4
  for (int kf = 0; kf < 32; ++kf) {
    int c0 = kf * 32 + c0sel;
    v16bf a;
#pragma unroll
    for (int j = 0; j < 8; ++j) {
      a[j]     = (__bf16)xr[c0 + j];        // K = c0 .. c0+7   (VGPR 0-3)
      a[8 + j] = (__bf16)xr[c0 + 16 + j];   // K = c0+16..c0+23 (VGPR 4-7)
    }
    v16bf bfrag = __builtin_bit_cast(v16bf, wpack[(size_t)tile * 32 * 32 + kf * 32 + lane]);
    acc = __builtin_amdgcn_wmma_f32_16x16x32_bf16(false, a, false, bfrag,
                                                  (short)0, acc, false, false);
  }
  if (mat < 2) {
#pragma unroll
    for (int r = 0; r < 8; ++r) acc[r] = elu1(acc[r]);
  }
  int col = ntile * 16 + (lane & 15);
  float* dst = (mat == 0) ? Qb : (mat == 1) ? Kb : Vb;
  dst += col;
#pragma unroll
  for (int r = 0; r < 8; ++r) {
    int rr = mtile * 16 + r + ((lane < 16) ? 0 : 8);
    dst[(size_t)rr * DH] = acc[r];
  }
}

// ---------------------------------------------------------------------------
// 3. Per-(batch,chunk) additive state delta. State layout (896 floats):
//    [0,256) leaf[n][i][j]; [256,832) Ws by level; [832,896) z.
// ---------------------------------------------------------------------------
__global__ void hss_deltaKernel(const float* __restrict__ Kg,
                                const float* __restrict__ Vg,
                                Bases B, float* __restrict__ delta) {
  __shared__ float sUV[2 * 464];        // UL | VR
  __shared__ float sA[CHUNK * NSLOT];   // a_t = UL^T k_top
  __shared__ float sBv[CHUNK * NSLOT];  // b_t = VR^T v_bot
  int tid = threadIdx.x;
  int b = blockIdx.x >> 5, c = blockIdx.x & 31;
  int rowBase = b * TLEN + c * CHUNK;
  __builtin_prefetch(Kg + (size_t)rowBase * DH, 0, 0);
  __builtin_prefetch(Vg + (size_t)rowBase * DH, 0, 0);

  for (int i = tid; i < 2 * 464; i += 256) {
    int m = i / 464, p = i % 464, l = levelOfBase(p);
    sUV[i] = B.p[(m == 0 ? 0 : 4) + l][p - baseOff(l)];
  }
  __syncthreads();

  for (int i = tid; i < CHUNK * NSLOT; i += 256) {
    int t = i / NSLOT, s = i % NSLOT;
    int l = levelOfSlot(s), rem = s - slotOff(l);
    int rl = RLf(l), bs_ = BSf(l);
    int n = rem / rl, r = rem % rl;
    const float* krow = Kg + (size_t)(rowBase + t) * DH + n * 2 * bs_;
    const float* vrow = Vg + (size_t)(rowBase + t) * DH + n * 2 * bs_ + bs_;
    float sa = 0.f, sb = 0.f;
    for (int d = 0; d < bs_; ++d) {
      sa += sUV[baseOff(l) + d * rl + r] * krow[d];
      sb += sUV[464 + baseOff(l) + d * rl + r] * vrow[d];
    }
    sA[t * NSLOT + s] = sa;
    sBv[t * NSLOT + s] = sb;
  }
  __syncthreads();

  for (int e = tid; e < STATE_N; e += 256) {
    float s = 0.f;
    if (e < 256) {
      int n = e >> 4, i = (e >> 2) & 3, j = e & 3;
      for (int t = 0; t < CHUNK; ++t)
        s += Kg[(size_t)(rowBase + t) * DH + 4 * n + i] *
             Vg[(size_t)(rowBase + t) * DH + 4 * n + j];
    } else if (e < 832) {
      int l = wLevel(e), rem = e - wOff(l), rl = RLf(l);
      int n = rem / (rl * rl), ij = rem % (rl * rl), i = ij / rl, j = ij % rl;
      int sa = slotOff(l) + n * rl + i, sb = slotOff(l) + n * rl + j;
      for (int t = 0; t < CHUNK; ++t)
        s += sA[t * NSLOT + sa] * sBv[t * NSLOT + sb];
    } else {
      int dd = e - 832;
      for (int t = 0; t < CHUNK; ++t)
        s += Kg[(size_t)(rowBase + t) * DH + dd];
    }
    delta[(size_t)(b * NCHUNK + c) * STATE_N + e] = s;
  }
}

// ---------------------------------------------------------------------------
// 4. Exclusive prefix-sum over chunks (per batch, per state element).
// ---------------------------------------------------------------------------
__global__ void hss_prefixKernel(const float* __restrict__ delta,
                                 float* __restrict__ start) {
  int b = blockIdx.x;
  int e = blockIdx.y * 256 + threadIdx.x;
  if (e >= STATE_N) return;
  float run = 0.f;
  for (int c = 0; c < NCHUNK; ++c) {
    size_t idx = (size_t)(b * NCHUNK + c) * STATE_N + e;
    start[idx] = run;
    run += delta[idx];
  }
}

// ---------------------------------------------------------------------------
// 5. Per-chunk exact replay of the recurrence from the prefix state.
// ---------------------------------------------------------------------------
__global__ void hss_scanKernel(const float* __restrict__ Qg,
                               const float* __restrict__ Kg,
                               const float* __restrict__ Vg,
                               Bases B, const float* __restrict__ start,
                               float* __restrict__ Yg) {
  __shared__ float sQ[CHUNK * DH], sK[CHUNK * DH], sV[CHUNK * DH];
  __shared__ float sBase[4 * 464];     // UL | VR | UR | VL
  __shared__ float sState[STATE_N];
  __shared__ float sProj[4 * NSLOT];   // a | b | c | d
  __shared__ float sEF[2 * NSLOT];     // e = Ws c | f = Ws^T d
  __shared__ float sY[DH];
  __shared__ float sRed[DH];
  __shared__ float sDen;
  int tid = threadIdx.x;
  int b = blockIdx.x >> 5, c = blockIdx.x & 31;
  size_t rowBase = (size_t)(b * TLEN + c * CHUNK);
  __builtin_prefetch(Qg + rowBase * DH, 0, 0);
  __builtin_prefetch(Kg + rowBase * DH, 0, 0);

  for (int i = tid; i < CHUNK * DH; i += 256) {
    sQ[i] = Qg[rowBase * DH + i];
    sK[i] = Kg[rowBase * DH + i];
    sV[i] = Vg[rowBase * DH + i];
  }
  for (int i = tid; i < 4 * 464; i += 256) {
    int m = i / 464, p = i % 464, l = levelOfBase(p);
    sBase[i] = B.p[m * 4 + l][p - baseOff(l)];
  }
  for (int e = tid; e < STATE_N; e += 256)
    sState[e] = start[(size_t)(b * NCHUNK + c) * STATE_N + e];
  __syncthreads();

  for (int t = 0; t < CHUNK; ++t) {
    const float* kt = sK + t * DH;
    const float* vt = sV + t * DH;
    const float* qt = sQ + t * DH;

    // (i) projections a,b,c,d  (352 tasks)
    if (tid < 4 * NSLOT) {
      int m = tid / NSLOT, s = tid % NSLOT;
      int l = levelOfSlot(s), rem = s - slotOff(l);
      int rl = RLf(l), bs_ = BSf(l);
      int n = rem / rl, r = rem % rl;
      int bm = (m == 0) ? 0 : (m == 3) ? 3 : 1;        // a:UL, b:VR, c:VR, d:VL
      const float* bas = sBase + bm * 464 + baseOff(l);
      const float* vec = (m == 0) ? (kt + n * 2 * bs_)
                       : (m == 1) ? (vt + n * 2 * bs_ + bs_)
                       : (m == 2) ? (qt + n * 2 * bs_ + bs_)
                                  : (qt + n * 2 * bs_);
      float sum = 0.f;
      for (int d = 0; d < bs_; ++d) sum += bas[d * rl + r] * vec[d];
      sProj[tid] = sum;
    }
    __syncthreads();

    // (ii) state update: leaf += k v^T, Ws += a b^T, z += k
    for (int e = tid; e < STATE_N; e += 256) {
      if (e < 256) {
        int n = e >> 4, i = (e >> 2) & 3, j = e & 3;
        sState[e] += kt[4 * n + i] * vt[4 * n + j];
      } else if (e < 832) {
        int l = wLevel(e), rem = e - wOff(l), rl = RLf(l);
        int n = rem / (rl * rl), ij = rem % (rl * rl), i = ij / rl, j = ij % rl;
        sState[e] += sProj[slotOff(l) + n * rl + i] *
                     sProj[NSLOT + slotOff(l) + n * rl + j];
      } else {
        sState[e] += kt[e - 832];
      }
    }
    __syncthreads();

    // (iii) e = Ws c, f = Ws^T d (176 tasks) + denom partials
    if (tid < 2 * NSLOT) {
      int m = tid / NSLOT, s = tid % NSLOT;
      int l = levelOfSlot(s), rem = s - slotOff(l), rl = RLf(l);
      int n = rem / rl, r = rem % rl;
      const float* W  = sState + wOff(l) + n * rl * rl;
      const float* pr = sProj + (m == 0 ? 2 : 3) * NSLOT + slotOff(l) + n * rl;
      float sum = 0.f;
      if (m == 0) { for (int j = 0; j < rl; ++j) sum += W[r * rl + j] * pr[j]; }
      else        { for (int j = 0; j < rl; ++j) sum += W[j * rl + r] * pr[j]; }
      sEF[tid] = sum;
    } else if (tid < 2 * NSLOT + DH) {
      int dd = tid - 2 * NSLOT;
      sRed[dd] = sState[832 + dd] * qt[dd];
    }
    __syncthreads();

    // (iv) y numerator (64 dims) + denominator reduce
    if (tid < DH) {
      int dd = tid, n0 = dd >> 2, i0 = dd & 3;
      float y = 0.f;
      const float* L = sState + n0 * 16 + i0 * 4;
      for (int j = 0; j < 4; ++j) y += L[j] * qt[4 * n0 + j];
      for (int l = 0; l < 4; ++l) {
        int bs_ = BSf(l), rl = RLf(l);
        int n = dd / (2 * bs_), pos = dd % (2 * bs_);
        if (pos < bs_) {                       // y_top = UL (Ws (VR^T q_bot))
          const float* bas = sBase + baseOff(l) + pos * rl;          // UL
          const float* e_  = sEF + slotOff(l) + n * rl;
          for (int r = 0; r < rl; ++r) y += bas[r] * e_[r];
        } else {                               // y_bot = UR (Ws^T (VL^T q_top))
          const float* bas = sBase + 2 * 464 + baseOff(l) + (pos - bs_) * rl; // UR
          const float* f_  = sEF + NSLOT + slotOff(l) + n * rl;
          for (int r = 0; r < rl; ++r) y += bas[r] * f_[r];
        }
      }
      sY[dd] = y;
    } else if (tid == DH) {
      float s = 0.f;
      for (int i = 0; i < DH; ++i) s += sRed[i];
      sDen = fmaxf(s, 1e-6f);
    }
    __syncthreads();

    // (v) normalize and emit
    if (tid < DH) Yg[(rowBase + t) * DH + tid] = sY[tid] / sDen;
    __syncthreads();
  }
}

// ---------------------------------------------------------------------------
// 6. out = Y @ Wo with exact-fp32 v_wmma_f32_16x16x4_f32. 4 waves/block.
// ---------------------------------------------------------------------------
__global__ void hss_outKernel(const float* __restrict__ Yb,
                              const float* __restrict__ Wo,
                              float* __restrict__ out) {
  int lane  = threadIdx.x & 31;
  int wid   = threadIdx.x >> 5;
  int mtile = blockIdx.x;                 // 0..1023
  int ntile = blockIdx.y * 4 + wid;       // 0..63
  int mrow  = mtile * 16 + (lane & 15);
  int col   = ntile * 16 + (lane & 15);
  int ksel  = (lane < 16) ? 0 : 2;
  const float* yr = Yb + (size_t)mrow * DH;
  v8f acc = {};
#pragma unroll
  for (int kf = 0; kf < 16; ++kf) {
    int k0 = kf * 4 + ksel;
    v2f a = { yr[k0], yr[k0 + 1] };
    v2f bvec = { Wo[(size_t)k0 * DMODEL + col],
                 Wo[(size_t)(k0 + 1) * DMODEL + col] };
    acc = __builtin_amdgcn_wmma_f32_16x16x4_f32(false, a, false, bvec,
                                                (short)0, acc, false, false);
  }
#pragma unroll
  for (int r = 0; r < 8; ++r) {
    int rr = mtile * 16 + r + ((lane < 16) ? 0 : 8);
    out[(size_t)rr * DMODEL + col] = acc[r];
  }
}

// ---------------------------------------------------------------------------
// launch
// ---------------------------------------------------------------------------
extern "C" void kernel_launch(void* const* d_in, const int* in_sizes, int n_in,
                              void* d_out, int out_size, void* d_ws, size_t ws_size,
                              hipStream_t stream) {
  const float* x  = (const float*)d_in[0];
  const float* Wq = (const float*)d_in[1];
  const float* Wk = (const float*)d_in[2];
  const float* Wv = (const float*)d_in[3];
  const float* Wo = (const float*)d_in[4];
  Bases bases;
  for (int i = 0; i < 16; ++i) bases.p[i] = (const float*)d_in[5 + i];

  // workspace layout (floats)
  float* ws = (float*)d_ws;
  const size_t WPACK_F = 98304;                 // 12288 frag-lanes * 8 dwords
  const size_t MAT_F   = (size_t)BATCH * TLEN * DH;  // 1,048,576
  const size_t ST_F    = (size_t)BATCH * NCHUNK * STATE_N;
  v8u* wpack   = (v8u*)d_ws;
  float* Qb    = ws + WPACK_F;
  float* Kb    = Qb + MAT_F;
  float* Vb    = Kb + MAT_F;
  float* Yb    = Vb + MAT_F;
  float* dlt   = Yb + MAT_F;
  float* stArr = dlt + ST_F;

  hss_packKernel<<<48, 256, 0, stream>>>(Wq, Wk, Wv, wpack);
  hss_qkvKernel<<<dim3(1024, 12), 32, 0, stream>>>(x, wpack, Qb, Kb, Vb);
  hss_deltaKernel<<<BATCH * NCHUNK, 256, 0, stream>>>(Kb, Vb, bases, dlt);
  hss_prefixKernel<<<dim3(BATCH, 4), 256, 0, stream>>>(dlt, stArr);
  hss_scanKernel<<<BATCH * NCHUNK, 256, 0, stream>>>(Qb, Kb, Vb, bases, stArr, Yb);
  hss_outKernel<<<dim3(1024, 16), 128, 0, stream>>>(Yb, Wo, (float*)d_out);
}
